// RecurrentModel_64201171141030
// MI455X (gfx1250) — matrix operand order repaired and checked by reading, not compiled, not run on hardware
//
#include <hip/hip_runtime.h>
#include <cmath>

typedef __attribute__((ext_vector_type(16))) _Float16 v16h;
typedef __attribute__((ext_vector_type(8)))  _Float16 v8h;
typedef __attribute__((ext_vector_type(8)))  float    v8f;
typedef __attribute__((ext_vector_type(4)))  float    v4f;

#if defined(__has_builtin)
#if __has_builtin(__builtin_amdgcn_tanhf)
#define FAST_TANH(x) __builtin_amdgcn_tanhf(x)
#endif
#endif
#ifndef FAST_TANH
#define FAST_TANH(x) tanhf(x)
#endif

#define HID   256      // hidden size
#define TLEN  256      // timesteps
#define VOUT  10       // output classes
#define BT    16       // batch rows per workgroup
#define NWAVE 8        // waves per workgroup; each wave owns TWO 16x16 column tiles
#define BLOCK (NWAVE * 32)
#define AROW  264      // padded row length (halves): 264*2 = 528 B = 33*16 -> conflict-free b128
#define HROW  (HID + 4)

// Split workgroup barrier: publish (wait for own DS ops, signal), then do
// independent work, then wait for the rest of the workgroup.
#define BAR_PUBLISH() asm volatile("s_wait_dscnt 0x0\n\ts_barrier_signal -1" ::: "memory")
#define BAR_ARRIVE()  asm volatile("s_barrier_wait -1" ::: "memory")

__global__ __launch_bounds__(BLOCK, 2)
void rnn_scan_kernel(const float* __restrict__ x,
                     const float* __restrict__ W_e, const float* __restrict__ b_e,
                     const float* __restrict__ W_h, const float* __restrict__ b_h,
                     const float* __restrict__ W_o, const float* __restrict__ b_o,
                     float* __restrict__ out)
{
    // ping-pong A-operand buffers -> one barrier per timestep
    __shared__ __align__(16) _Float16 a_lds[2][BT * AROW];
    __shared__ __align__(16) float xbufT[(TLEN + 1) * BT]; // x transposed [t][r], +1 pad row
    __shared__ float hbuf[BT * HROW];                      // final hidden state (f32)

    const int tid = threadIdx.x;
    const int wid = tid >> 5;          // wave id: owns columns [16w,16w+16) and +128
    const int tl  = tid & 31;          // lane in wave (wave32)
    const int lo  = tl & 15;
    const bool hi = (tl >= 16);
    const int r0  = blockIdx.x * BT;   // first batch row of this workgroup

    // ---- stage x[r0:r0+16, :] transposed into LDS: xbufT[t*16 + r] ----
    for (int i = tid; i < BT * TLEN; i += BLOCK) {
        int r = i >> 8, t = i & (TLEN - 1);        // read coalesced along t
        xbufT[t * BT + r] = x[(size_t)(r0 + r) * TLEN + t];
    }

    // ---- per-lane column constants for both tiles (fixed for the whole scan) ----
    const int   c0    = wid * 16 + lo;
    const int   c1    = c0 + 128;
    const float we_c0 = W_e[c0], we_c1 = W_e[c1];
    const float be_c0 = b_e[c0], be_c1 = b_e[c1];
    const float bh_c0 = b_h[c0], bh_c1 = b_h[c1];

    // ---- preload B fragments of W_h into VGPRs (f16) for BOTH column tiles ----
    // 16-bit B 32x16 layout: lanes 0-15 hold col N=lane, K = 0..7,16..23 packed 2/VGPR;
    // lanes 16-31 hold same cols, K = 8..15,24..31.
    v16h bfrag0[8], bfrag1[8];
    {
        const int kbase = hi ? 8 : 0;
        #pragma unroll
        for (int kb = 0; kb < 8; ++kb) {
            #pragma unroll
            for (int j = 0; j < 16; ++j) {
                const size_t row = (size_t)(kb * 32 + kbase + j + (j >= 8 ? 8 : 0)) * HID;
                bfrag0[kb][j] = (_Float16)W_h[row + c0];
                bfrag1[kb][j] = (_Float16)W_h[row + c1];
            }
        }
    }

    __syncthreads();

    // ---- t = 0 operand: a = tanh(x[:,0] * W_e + b_e)  (h0 == 0) ----
    for (int i = tid; i < BT * HID; i += BLOCK) {
        int r = i >> 8, c = i & (HID - 1);
        a_lds[0][r * AROW + c] =
            (_Float16)FAST_TANH(xbufT[0 * BT + r] * W_e[c] + b_e[c]);
    }
    __syncthreads();

    // A-fragment addressing (16-bit A 16x32 layout):
    // lanes 0-15: row = lane, halves 0..7 = K 0..7,  halves 8..15 = K 16..23
    // lanes16-31: row = lane-16, halves 0..7 = K 8..15, halves 8..15 = K 24..31
    const int arow  = lo;
    const int abase = hi ? 8 : 0;
    const int rb    = hi ? 8 : 0;      // C-matrix row base for this half-wave

    // One timestep GEMM for both tiles. Each A fragment feeds two WMMAs
    // (tiles share rows), halving LDS traffic; one accumulator chain per tile
    // gives RAW distance 2 between dependent WMMAs.
    #define STEP_GEMM2(SRC, ACC0, ACC1)                                         \
        {                                                                       \
            _Pragma("unroll")                                                   \
            for (int kb = 0; kb < 8; ++kb) {                                    \
                const _Float16* p0 = &(SRC)[arow * AROW + kb * 32 + abase];     \
                v8h a0 = *(const v8h*)(p0);                                     \
                v8h a1 = *(const v8h*)(p0 + 16);                                \
                v16h aA = __builtin_shufflevector(a0, a1,                       \
                            0,1,2,3,4,5,6,7,8,9,10,11,12,13,14,15);             \
                ACC0 = __builtin_amdgcn_wmma_f32_16x16x32_f16(                  \
                           false, aA, false, bfrag0[kb], (short)0, ACC0,        \
                           false, false);                                       \
                ACC1 = __builtin_amdgcn_wmma_f32_16x16x32_f16(                  \
                           false, aA, false, bfrag1[kb], (short)0, ACC1,        \
                           false, false);                                       \
            }                                                                   \
        }

    // next-step input embedding for this thread's 8 rows x 2 columns
    // (2 b128 broadcast loads + 16 independent fma/tanh chains)
    #define EMBED16(T, INP0, INP1)                                              \
        {                                                                       \
            const v4f* xp = (const v4f*)&xbufT[(T) * BT + rb];                  \
            v4f xv0 = xp[0];                                                    \
            v4f xv1 = xp[1];                                                    \
            _Pragma("unroll")                                                   \
            for (int j = 0; j < 4; ++j) {                                       \
                (INP0)[j]     = FAST_TANH(xv0[j] * we_c0 + be_c0);              \
                (INP0)[4 + j] = FAST_TANH(xv1[j] * we_c0 + be_c0);              \
                (INP1)[j]     = FAST_TANH(xv0[j] * we_c1 + be_c1);              \
                (INP1)[4 + j] = FAST_TANH(xv1[j] * we_c1 + be_c1);              \
            }                                                                   \
        }

    float inp0[8], inp1[8];
    EMBED16(1, inp0, inp1);    // embedding consumed by step t = 0's epilogue

    int p = 0;
    for (int t = 0; t + 1 < TLEN; ++t) {
        v8f acc0 = {}, acc1 = {};
        const _Float16* src = a_lds[p];
        STEP_GEMM2(src, acc0, acc1);

        // epilogue into the *other* buffer: 16 independent tanh chains,
        // then 16 f16 stores (rows j+rb, columns c0 / c1).
        _Float16* dst = a_lds[p ^ 1];
        float hv0[8], hv1[8];
        #pragma unroll
        for (int j = 0; j < 8; ++j) {
            hv0[j] = FAST_TANH(acc0[j] + bh_c0);
            hv1[j] = FAST_TANH(acc1[j] + bh_c1);
        }
        #pragma unroll
        for (int j = 0; j < 8; ++j) {
            int r = j + rb;
            dst[r * AROW + c0] = (_Float16)(hv0[j] + inp0[j]);
            dst[r * AROW + c1] = (_Float16)(hv1[j] + inp1[j]);
        }

        // split barrier: publish our stores, then hide the next-next-step
        // embedding (independent of the ping-pong buffers) in the wait window.
        BAR_PUBLISH();
        EMBED16(t + 2, inp0, inp1);   // at t+2 == TLEN this reads the pad row (unused)
        BAR_ARRIVE();
        p ^= 1;
    }

    // ---- peeled final step: keep full-precision h for the output projection ----
    {
        v8f acc0 = {}, acc1 = {};
        const _Float16* src = a_lds[p];
        STEP_GEMM2(src, acc0, acc1);
        #pragma unroll
        for (int j = 0; j < 8; ++j) {
            int r = j + rb;
            hbuf[r * HROW + c0] = FAST_TANH(acc0[j] + bh_c0);
            hbuf[r * HROW + c1] = FAST_TANH(acc1[j] + bh_c1);
        }
        __syncthreads();
    }

    // ---- output projection: out[r, v] = h[r,:] . W_o[:, v] + b_o[v] ----
    if (tid < BT * VOUT) {
        int r = tid / VOUT, v = tid - r * VOUT;
        float s = b_o[v];
        for (int c = 0; c < HID; ++c)
            s += hbuf[r * HROW + c] * W_o[c * VOUT + v];
        out[(size_t)(r0 + r) * VOUT + v] = s;
    }
}

extern "C" void kernel_launch(void* const* d_in, const int* in_sizes, int n_in,
                              void* d_out, int out_size, void* d_ws, size_t ws_size,
                              hipStream_t stream) {
    const float* x   = (const float*)d_in[0];
    const float* W_e = (const float*)d_in[1];
    const float* b_e = (const float*)d_in[2];
    const float* W_h = (const float*)d_in[3];
    const float* b_h = (const float*)d_in[4];
    const float* W_o = (const float*)d_in[5];
    const float* b_o = (const float*)d_in[6];
    float* out = (float*)d_out;

    const int B = in_sizes[0] / TLEN;      // x is [B, T, 1]
    const int grid = B / BT;               // 2048/16 = 128 workgroups

    rnn_scan_kernel<<<grid, BLOCK, 0, stream>>>(x, W_e, b_e, W_h, b_h, W_o, b_o, out);
}
